// ST_Transformer_adaptive_515396075928
// MI455X (gfx1250) — matrix operand order repaired and checked by reading, not compiled, not run on hardware
//
#include <hip/hip_runtime.h>
#include <math.h>
#include <stdint.h>

// ---------------- problem constants (match reference) ----------------
#define NN    8192
#define EE    131072
#define D_IN  3000
#define FH1_  256
#define FH2_  64
#define GH1_  128
#define GH2_  64
#define N_E_  128
#define E_DIM 64
#define KCL   10
#define BNSCALE 0.9999500037496876f   // 1/sqrt(1+1e-4)

// ---------------- WMMA types ----------------
typedef __attribute__((ext_vector_type(16))) __bf16 v16bf;
typedef __attribute__((ext_vector_type(8)))  float  v8f;

union BfVec { v16bf v; unsigned short u[16]; uint4 q[2]; };

__device__ __forceinline__ unsigned short f2bf(float f) {
  unsigned int b = __float_as_uint(f);
  b += 0x7FFFu + ((b >> 16) & 1u);          // round to nearest even
  return (unsigned short)(b >> 16);
}

__device__ __forceinline__ v16bf load_frag(const unsigned short* p) {
  BfVec x;
  x.q[0] = *(const uint4*)(p);
  x.q[1] = *(const uint4*)(p + 8);
  return x.v;
}

// ======================================================================
// Weight pre-pack: B[K,N] fp32 -> bf16 WMMA B-fragments.
// Fragment (kt,nt) = 32 lanes x 16 ushort, contiguous per lane:
//   Bp[((nt*Kt + kt)*32 + lane)*16 + i] = bf16(B[kt*32 + (lane>>4)*16 + i][nt*16 + (lane&15)])
// Out-of-range K/N entries are zero-filled (removes all guards from GEMM).
// ======================================================================
__global__ void pack_b(const float* __restrict__ B, unsigned short* __restrict__ Bp,
                       int K, int N) {
  int idx  = blockIdx.x * 256 + threadIdx.x;
  int lane = idx & 31;
  int tile = idx >> 5;
  int Kt = (K + 31) >> 5, Nt = (N + 15) >> 4;
  if (tile >= Kt * Nt) return;
  int nt = tile / Kt, kt = tile % Kt;
  int n  = nt * 16 + (lane & 15);
  int kb = kt * 32 + (lane >> 4) * 16;
  unsigned short* o = Bp + ((size_t)tile * 32 + lane) * 16;
  #pragma unroll
  for (int i = 0; i < 16; ++i) {
    int k = kb + i;
    o[i] = (k < K && n < N) ? f2bf(B[(size_t)k * N + n]) : (unsigned short)0;
  }
}

// ======================================================================
// Tiled GEMM:  C[M,N] = A[M,K] * B[K,N]  (A fp32->bf16 on the fly,
// B pre-packed bf16 fragments, fp32 accumulate).
// block = 256 = 8 waves; wave w -> rows [bx*128+16w, +16),
// cols [by*64, +64) as 4 WMMA accumulators reusing one A fragment.
// mode 0: plain   1: +bias   2: +bias, *bn_g*BNSCALE, +bn_b, ELU
// ======================================================================
__global__ __launch_bounds__(256) void gemm_bf16_wmma(
    const float* __restrict__ A, const unsigned short* __restrict__ Bp,
    float* __restrict__ C, int K, int N,
    const float* __restrict__ bias, const float* __restrict__ gamma,
    const float* __restrict__ beta, int mode)
{
  const int wave = threadIdx.x >> 5;
  const int lane = threadIdx.x & 31;
  const int half = lane >> 4;          // 0: lanes 0-15, 1: lanes 16-31
  const int l15  = lane & 15;
  const int rowBase = blockIdx.x * 128 + wave * 16;
  const int colBase = blockIdx.y * 64;
  const int Kt = (K + 31) >> 5;

  const float* arow = A + (size_t)(rowBase + l15) * K;

  // per-lane packed-B pointers for the 4 column sub-tiles (512 ushort / fragment)
  const int nt0 = colBase >> 4;
  const unsigned short* bp0 = Bp + ((size_t)(nt0 + 0) * Kt * 32 + lane) * 16;
  const unsigned short* bp1 = Bp + ((size_t)(nt0 + 1) * Kt * 32 + lane) * 16;
  const unsigned short* bp2 = Bp + ((size_t)(nt0 + 2) * Kt * 32 + lane) * 16;
  const unsigned short* bp3 = Bp + ((size_t)(nt0 + 3) * Kt * 32 + lane) * 16;

  v8f acc0 = {}, acc1 = {}, acc2 = {}, acc3 = {};

  const int KtMain = K >> 5;           // full 32-wide K tiles
  int kt = 0;
  for (; kt < KtMain; ++kt) {
    // ---- A fragment (guard-free): lane = row, halves hold K+0..15 / K+16..31
    const int kb = (kt << 5) + half * 16;
    float4 f0 = *(const float4*)(arow + kb);
    float4 f1 = *(const float4*)(arow + kb + 4);
    float4 f2 = *(const float4*)(arow + kb + 8);
    float4 f3 = *(const float4*)(arow + kb + 12);
    BfVec a;
    a.u[0]=f2bf(f0.x);  a.u[1]=f2bf(f0.y);  a.u[2]=f2bf(f0.z);  a.u[3]=f2bf(f0.w);
    a.u[4]=f2bf(f1.x);  a.u[5]=f2bf(f1.y);  a.u[6]=f2bf(f1.z);  a.u[7]=f2bf(f1.w);
    a.u[8]=f2bf(f2.x);  a.u[9]=f2bf(f2.y);  a.u[10]=f2bf(f2.z); a.u[11]=f2bf(f2.w);
    a.u[12]=f2bf(f3.x); a.u[13]=f2bf(f3.y); a.u[14]=f2bf(f3.z); a.u[15]=f2bf(f3.w);

    const size_t ko = (size_t)kt * 512;
    v16bf b0 = load_frag(bp0 + ko);
    v16bf b1 = load_frag(bp1 + ko);
    v16bf b2 = load_frag(bp2 + ko);
    v16bf b3 = load_frag(bp3 + ko);

    acc0 = __builtin_amdgcn_wmma_f32_16x16x32_bf16(false, a.v, false, b0, (short)0, acc0, false, false);
    acc1 = __builtin_amdgcn_wmma_f32_16x16x32_bf16(false, a.v, false, b1, (short)0, acc1, false, false);
    acc2 = __builtin_amdgcn_wmma_f32_16x16x32_bf16(false, a.v, false, b2, (short)0, acc2, false, false);
    acc3 = __builtin_amdgcn_wmma_f32_16x16x32_bf16(false, a.v, false, b3, (short)0, acc3, false, false);
  }

  if (kt < Kt) {                       // K tail (only K=3000): guarded A, B is zero-padded
    const int kb = (kt << 5) + half * 16;
    BfVec a;
    #pragma unroll
    for (int i = 0; i < 8; ++i) {
      int ka = kb + i;      a.u[i]     = (ka < K) ? f2bf(arow[ka]) : (unsigned short)0;
      int kc = kb + 8 + i; a.u[8 + i] = (kc < K) ? f2bf(arow[kc]) : (unsigned short)0;
    }
    const size_t ko = (size_t)kt * 512;
    v16bf b0 = load_frag(bp0 + ko);
    v16bf b1 = load_frag(bp1 + ko);
    v16bf b2 = load_frag(bp2 + ko);
    v16bf b3 = load_frag(bp3 + ko);
    acc0 = __builtin_amdgcn_wmma_f32_16x16x32_bf16(false, a.v, false, b0, (short)0, acc0, false, false);
    acc1 = __builtin_amdgcn_wmma_f32_16x16x32_bf16(false, a.v, false, b1, (short)0, acc1, false, false);
    acc2 = __builtin_amdgcn_wmma_f32_16x16x32_bf16(false, a.v, false, b2, (short)0, acc2, false, false);
    acc3 = __builtin_amdgcn_wmma_f32_16x16x32_bf16(false, a.v, false, b3, (short)0, acc3, false, false);
  }

  // ---- epilogue: C/D layout lanes 0-15: M=r ; lanes 16-31: M=r+8 ----
  const int rOff = half * 8;
#define EPILOGUE(T, ACC)                                                     \
  {                                                                          \
    const int col = colBase + 16 * (T) + l15;                                \
    if (col < N) {                                                           \
      float bi = (mode >= 1) ? bias[col] : 0.f;                              \
      float ga = 1.f, be = 0.f;                                              \
      if (mode == 2) { ga = gamma[col] * BNSCALE; be = beta[col]; }          \
      _Pragma("unroll")                                                      \
      for (int r = 0; r < 8; ++r) {                                          \
        float val = ACC[r];                                                  \
        if (mode == 1) val += bi;                                            \
        else if (mode == 2) {                                                \
          val = (val + bi) * ga + be;                                        \
          val = val > 0.f ? val : (__expf(val) - 1.f);                       \
        }                                                                    \
        C[(size_t)(rowBase + rOff + r) * N + col] = val;                     \
      }                                                                      \
    }                                                                        \
  }
  EPILOGUE(0, acc0)
  EPILOGUE(1, acc1)
  EPILOGUE(2, acc2)
  EPILOGUE(3, acc3)
#undef EPILOGUE
}

// ======================================================================
// Edge-parallel attention kernels (wave32-aware)
// ======================================================================
__device__ __forceinline__ void atomicMaxFloat(float* addr, float val) {
  if (val >= 0.f) atomicMax((int*)addr, __float_as_int(val));
  else            atomicMin((unsigned int*)addr, __float_as_uint(val));
}

__global__ void init_maxsum(float* mx, float* sm, int n) {
  int i = blockIdx.x * blockDim.x + threadIdx.x;
  if (i < n) { mx[i] = -INFINITY; sm[i] = 0.f; }
}

__global__ void zero_buf(float* p, int n) {
  int i = blockIdx.x * blockDim.x + threadIdx.x;
  if (i < n) p[i] = 0.f;
}

__global__ void relu_buf(float* p, int n) {
  int i = blockIdx.x * blockDim.x + threadIdx.x;
  if (i < n) p[i] = fmaxf(p[i], 0.f);
}

// wave per edge: logit = <q[dst],k[src]>/sqrt(d); segment max over dst
__global__ __launch_bounds__(256) void edge_logits(
    const float* q, const float* k, const int* src, const int* dst,
    float* logit, float* mx, int d, float scale)
{
  int e = blockIdx.x * 8 + (threadIdx.x >> 5);
  if (e >= EE) return;
  int lane = threadIdx.x & 31;
  const float* qp = q + (size_t)dst[e] * d;
  const float* kp = k + (size_t)src[e] * d;
  float p = 0.f;
  for (int j = lane; j < d; j += 32) p += qp[j] * kp[j];
  for (int off = 16; off; off >>= 1) p += __shfl_xor(p, off, 32);
  if (lane == 0) {
    float lg = p * scale;
    logit[e] = lg;
    atomicMaxFloat(mx + dst[e], lg);
  }
}

// thread per edge: ex = exp(logit - max[dst]); segment sum
__global__ void edge_exp(float* buf, const int* dst, const float* mx, float* sm) {
  int e = blockIdx.x * blockDim.x + threadIdx.x;
  if (e >= EE) return;
  int t = dst[e];
  float v = __expf(buf[e] - mx[t]);
  buf[e] = v;
  atomicAdd(sm + t, v);
}

// wave per edge: alpha = ex/sum[dst]; agg[dst] += alpha * v[src]
__global__ __launch_bounds__(256) void edge_alpha_agg(
    float* buf, const float* sm, const int* src, const int* dst,
    const float* v, float* agg, int d)
{
  int e = blockIdx.x * 8 + (threadIdx.x >> 5);
  if (e >= EE) return;
  int lane = threadIdx.x & 31;
  int s = src[e], t = dst[e];
  float a = buf[e] / sm[t];        // in-wave load-before-store ordering is preserved
  if (lane == 0) buf[e] = a;       // buf becomes alpha
  const float* vp = v + (size_t)s * d;
  float* op = agg + (size_t)t * d;
  for (int j = lane; j < d; j += 32) atomicAdd(op + j, a * vp[j]);
}

// dense(ei,alpha) @ h : out[src] += w * alpha[e] * h[dst]
__global__ __launch_bounds__(256) void edge_combine(
    const float* alpha, const int* src, const int* dst,
    const float* h, float* out, int d, float w)
{
  int e = blockIdx.x * 8 + (threadIdx.x >> 5);
  if (e >= EE) return;
  int lane = threadIdx.x & 31;
  float a = w * alpha[e];
  const float* hp = h + (size_t)dst[e] * d;
  float* op = out + (size_t)src[e] * d;
  for (int j = lane; j < d; j += 32) atomicAdd(op + j, a * hp[j]);
}

// ======================================================================
// VQ argmin + concat z = [cb[argmin], mu]   (wave per node)
// ======================================================================
__global__ __launch_bounds__(256) void vq_concat(
    const float* feat, const float* cb, const float* mu, float* z)
{
  __shared__ float sfeat[8][64];
  int wave = threadIdx.x >> 5, lane = threadIdx.x & 31;
  int node = blockIdx.x * 8 + wave;
  sfeat[wave][lane]      = feat[(size_t)node * 64 + lane];
  sfeat[wave][lane + 32] = feat[(size_t)node * 64 + lane + 32];
  __syncthreads();

  float best = INFINITY; int bi = 0;
  for (int c = lane; c < N_E_; c += 32) {
    const float* cp = cb + (size_t)c * 64;
    float s = 0.f;
    #pragma unroll 4
    for (int j = 0; j < 64; ++j) { float df = sfeat[wave][j] - cp[j]; s += df * df; }
    if (s < best) { best = s; bi = c; }
  }
  for (int off = 16; off; off >>= 1) {
    float ob = __shfl_xor(best, off, 32);
    int   oi = __shfl_xor(bi,   off, 32);
    if (ob < best || (ob == best && oi < bi)) { best = ob; bi = oi; }
  }
  const float* cp = cb + (size_t)bi * 64;
  float* zp = z + (size_t)node * 128;
  zp[lane]      = cp[lane];
  zp[lane + 32] = cp[lane + 32];
  zp[64 + lane] = mu[(size_t)node * 64 + lane];
  zp[96 + lane] = mu[(size_t)node * 64 + lane + 32];
}

// ======================================================================
// Student-t soft assignment (alpha=1 -> exponent 1), wave per node
// ======================================================================
__global__ __launch_bounds__(256) void student_t(
    const float* z, const float* cluster, float* qn)
{
  int wave = threadIdx.x >> 5, lane = threadIdx.x & 31;
  int node = blockIdx.x * 8 + wave;
  const float* zp = z + (size_t)node * 128;
  float qv = 0.f;
  if (lane < KCL) {
    const float* cp = cluster + (size_t)lane * 128;
    float s = 0.f;
    for (int j = 0; j < 128; ++j) { float df = zp[j] - cp[j]; s += df * df; }
    qv = 1.f / (1.f + s);
  }
  float tot = qv;
  for (int off = 16; off; off >>= 1) tot += __shfl_xor(tot, off, 32);
  if (lane < KCL) qn[(size_t)node * KCL + lane] = qv / tot;
}

// ======================================================================
// Host orchestration
// ======================================================================
static inline void launch_gemm(const float* A, const unsigned short* Bp, float* C,
                               int K, int Nc, const float* bias, const float* g,
                               const float* bt, int mode, hipStream_t s) {
  dim3 grid(NN / 128, (Nc + 63) / 64);
  gemm_bf16_wmma<<<grid, 256, 0, s>>>(A, Bp, C, K, Nc, bias, g, bt, mode);
}

static inline void edge_pass(const float* q, const float* k, const float* v,
                             const int* src, const int* dst,
                             float* alpha, float* agg, float* mx, float* sm,
                             int d, hipStream_t s) {
  init_maxsum<<<(NN + 255) / 256, 256, 0, s>>>(mx, sm, NN);
  edge_logits<<<EE / 8, 256, 0, s>>>(q, k, src, dst, alpha, mx, d, 1.0f / sqrtf((float)d));
  edge_exp<<<EE / 256, 256, 0, s>>>(alpha, dst, mx, sm);
  edge_alpha_agg<<<EE / 8, 256, 0, s>>>(alpha, sm, src, dst, v, agg, d);
}

extern "C" void kernel_launch(void* const* d_in, const int* in_sizes, int n_in,
                              void* d_out, int out_size, void* d_ws, size_t ws_size,
                              hipStream_t stream) {
  // ---- inputs (setup_inputs dict order, params flattened in insertion order)
  const float* x    = (const float*)d_in[0];
  const int*   adj  = (const int*)d_in[1];   // [0:E)=src, [E:2E)=dst
  const int*   adjp = (const int*)d_in[2];
  const float* enc1_W = (const float*)d_in[3];
  const float* enc1_b = (const float*)d_in[4];
  const float* bn1_g  = (const float*)d_in[5];
  const float* bn1_b  = (const float*)d_in[6];
  const float* enc2_W = (const float*)d_in[7];
  const float* enc2_b = (const float*)d_in[8];
  const float* bn2_g  = (const float*)d_in[9];
  const float* bn2_b  = (const float*)d_in[10];
  const float* gc1_Wq = (const float*)d_in[11]; const float* gc1_bq = (const float*)d_in[12];
  const float* gc1_Wk = (const float*)d_in[13]; const float* gc1_bk = (const float*)d_in[14];
  const float* gc1_Wv = (const float*)d_in[15]; const float* gc1_bv = (const float*)d_in[16];
  const float* gc1_Ws = (const float*)d_in[17]; const float* gc1_bs = (const float*)d_in[18];
  const float* c0_Wq = (const float*)d_in[19];  const float* c0_bq = (const float*)d_in[20];
  const float* c0_Wk = (const float*)d_in[21];  const float* c0_bk = (const float*)d_in[22];
  const float* c0_Wv = (const float*)d_in[23];  const float* c0_bv = (const float*)d_in[24];
  const float* c0_Ws = (const float*)d_in[25];  const float* c0_bs = (const float*)d_in[26];
  const float* gc2_Wq = (const float*)d_in[27]; const float* gc2_bq = (const float*)d_in[28];
  const float* gc2_Wk = (const float*)d_in[29]; const float* gc2_bk = (const float*)d_in[30];
  const float* gc2_Wv = (const float*)d_in[31]; const float* gc2_bv = (const float*)d_in[32];
  const float* gc2_Ws = (const float*)d_in[33]; const float* gc2_bs = (const float*)d_in[34];
  const float* dec_W = (const float*)d_in[35];  const float* dec_b = (const float*)d_in[36];
  const float* bnd_g = (const float*)d_in[37];  const float* bnd_b = (const float*)d_in[38];
  const float* codebook = (const float*)d_in[39];
  const float* cluster  = (const float*)d_in[40];

  const int* a_src = adj;        const int* a_dst = adj + EE;
  const int* p_src = adjp;       const int* p_dst = adjp + EE;

  // ---- outputs: z | de_feat | qn | feat  (flat, in return order)
  float* out    = (float*)d_out;
  float* z      = out;                                   // 8192*128
  float* defeat = z + (size_t)NN * 128;                  // 8192*3000
  float* qn     = defeat + (size_t)NN * D_IN;            // 8192*10
  float* feat   = qn + (size_t)NN * KCL;                 // 8192*64

  // ---- workspace carve (fp32 region)
  float* ws = (float*)d_ws;
  float* B_feat1 = ws;  ws += (size_t)NN * 256;
  float* B_q     = ws;  ws += (size_t)NN * 128;
  float* B_k     = ws;  ws += (size_t)NN * 128;
  float* B_v     = ws;  ws += (size_t)NN * 128;
  float* B_h1    = ws;  ws += (size_t)NN * 128;
  float* B_h1p   = ws;  ws += (size_t)NN * 128;
  float* B_comb  = ws;  ws += (size_t)NN * 128;
  float* B_mu    = ws;  ws += (size_t)NN * 64;
  float* B_al    = ws;  ws += EE;
  float* B_alp   = ws;  ws += EE;
  float* B_max   = ws;  ws += NN;
  float* B_sum   = ws;  ws += NN;

  // ---- bf16 packed-weight region (64B aligned)
  unsigned short* wsu = (unsigned short*)(((uintptr_t)ws + 63) & ~(uintptr_t)63);
  auto pack = [&](const float* W, int K, int Nc) -> const unsigned short* {
    int Kt = (K + 31) / 32, Nt = (Nc + 15) / 16;
    unsigned short* dst = wsu;
    wsu += (size_t)Kt * Nt * 512;
    int threads = Kt * Nt * 32;
    pack_b<<<(threads + 255) / 256, 256, 0, stream>>>(W, dst, K, Nc);
    return dst;
  };
  const unsigned short* P_enc1 = pack(enc1_W, D_IN, FH1_);
  const unsigned short* P_enc2 = pack(enc2_W, FH1_, FH2_);
  const unsigned short* P_g1q = pack(gc1_Wq, FH2_, GH1_);
  const unsigned short* P_g1k = pack(gc1_Wk, FH2_, GH1_);
  const unsigned short* P_g1v = pack(gc1_Wv, FH2_, GH1_);
  const unsigned short* P_g1s = pack(gc1_Ws, FH2_, GH1_);
  const unsigned short* P_c0q = pack(c0_Wq, GH1_, GH1_);
  const unsigned short* P_c0k = pack(c0_Wk, GH1_, GH1_);
  const unsigned short* P_c0v = pack(c0_Wv, GH1_, GH1_);
  const unsigned short* P_c0s = pack(c0_Ws, GH1_, GH1_);
  const unsigned short* P_g2q = pack(gc2_Wq, GH1_, GH2_);
  const unsigned short* P_g2k = pack(gc2_Wk, GH1_, GH2_);
  const unsigned short* P_g2v = pack(gc2_Wv, GH1_, GH2_);
  const unsigned short* P_g2s = pack(gc2_Ws, GH1_, GH2_);
  const unsigned short* P_dec = pack(dec_W, E_DIM + GH2_, D_IN);

  // ---- encoder
  launch_gemm(x, P_enc1, B_feat1, D_IN, FH1_, enc1_b, bn1_g, bn1_b, 2, stream);
  launch_gemm(B_feat1, P_enc2, feat, FH1_, FH2_, enc2_b, bn2_g, bn2_b, 2, stream);

  // ---- gc1 (q,k,v shared by both edge sets; skip GEMM seeds the agg buffer)
  launch_gemm(feat, P_g1q, B_q, FH2_, GH1_, gc1_bq, 0, 0, 1, stream);
  launch_gemm(feat, P_g1k, B_k, FH2_, GH1_, gc1_bk, 0, 0, 1, stream);
  launch_gemm(feat, P_g1v, B_v, FH2_, GH1_, gc1_bv, 0, 0, 1, stream);
  launch_gemm(feat, P_g1s, B_h1,  FH2_, GH1_, gc1_bs, 0, 0, 1, stream);
  launch_gemm(feat, P_g1s, B_h1p, FH2_, GH1_, gc1_bs, 0, 0, 1, stream);
  edge_pass(B_q, B_k, B_v, a_src, a_dst, B_al,  B_h1,  B_max, B_sum, GH1_, stream);
  edge_pass(B_q, B_k, B_v, p_src, p_dst, B_alp, B_h1p, B_max, B_sum, GH1_, stream);
  zero_buf<<<(NN * 128 + 255) / 256, 256, 0, stream>>>(B_comb, NN * 128);
  edge_combine<<<EE / 8, 256, 0, stream>>>(B_al,  a_src, a_dst, B_h1,  B_comb, GH1_, 0.5f);
  edge_combine<<<EE / 8, 256, 0, stream>>>(B_alp, p_src, p_dst, B_h1p, B_comb, GH1_, 0.5f);
  relu_buf<<<(NN * 128 + 255) / 256, 256, 0, stream>>>(B_comb, NN * 128);   // h

  // ---- conv0 first application: xn = tconv(h, ei)
  launch_gemm(B_comb, P_c0q, B_q, GH1_, GH1_, c0_bq, 0, 0, 1, stream);
  launch_gemm(B_comb, P_c0k, B_k, GH1_, GH1_, c0_bk, 0, 0, 1, stream);
  launch_gemm(B_comb, P_c0v, B_v, GH1_, GH1_, c0_bv, 0, 0, 1, stream);
  launch_gemm(B_comb, P_c0s, B_h1, GH1_, GH1_, c0_bs, 0, 0, 1, stream);
  edge_pass(B_q, B_k, B_v, a_src, a_dst, B_al, B_h1, B_max, B_sum, GH1_, stream);  // xn

  // ---- conv0 second application: xp = tconv(xn, eip)
  launch_gemm(B_h1, P_c0q, B_q, GH1_, GH1_, c0_bq, 0, 0, 1, stream);
  launch_gemm(B_h1, P_c0k, B_k, GH1_, GH1_, c0_bk, 0, 0, 1, stream);
  launch_gemm(B_h1, P_c0v, B_v, GH1_, GH1_, c0_bv, 0, 0, 1, stream);
  launch_gemm(B_h1, P_c0s, B_h1p, GH1_, GH1_, c0_bs, 0, 0, 1, stream);
  edge_pass(B_q, B_k, B_v, p_src, p_dst, B_alp, B_h1p, B_max, B_sum, GH1_, stream); // xp

  zero_buf<<<(NN * 128 + 255) / 256, 256, 0, stream>>>(B_comb, NN * 128);
  edge_combine<<<EE / 8, 256, 0, stream>>>(B_al,  a_src, a_dst, B_h1,  B_comb, GH1_, 0.5f);
  edge_combine<<<EE / 8, 256, 0, stream>>>(B_alp, p_src, p_dst, B_h1p, B_comb, GH1_, 0.5f);
  relu_buf<<<(NN * 128 + 255) / 256, 256, 0, stream>>>(B_comb, NN * 128);   // h2

  // ---- gc2 (q,k,v shared by both edge sets), d = 64
  launch_gemm(B_comb, P_g2q, B_q, GH1_, GH2_, gc2_bq, 0, 0, 1, stream);
  launch_gemm(B_comb, P_g2k, B_k, GH1_, GH2_, gc2_bk, 0, 0, 1, stream);
  launch_gemm(B_comb, P_g2v, B_v, GH1_, GH2_, gc2_bv, 0, 0, 1, stream);
  launch_gemm(B_comb, P_g2s, B_h1,  GH1_, GH2_, gc2_bs, 0, 0, 1, stream);
  launch_gemm(B_comb, P_g2s, B_h1p, GH1_, GH2_, gc2_bs, 0, 0, 1, stream);
  edge_pass(B_q, B_k, B_v, a_src, a_dst, B_al,  B_h1,  B_max, B_sum, GH2_, stream);
  edge_pass(B_q, B_k, B_v, p_src, p_dst, B_alp, B_h1p, B_max, B_sum, GH2_, stream);
  zero_buf<<<(NN * 64 + 255) / 256, 256, 0, stream>>>(B_mu, NN * 64);
  edge_combine<<<EE / 8, 256, 0, stream>>>(B_al,  a_src, a_dst, B_h1,  B_mu, GH2_, 0.5f);
  edge_combine<<<EE / 8, 256, 0, stream>>>(B_alp, p_src, p_dst, B_h1p, B_mu, GH2_, 0.5f);
  // (no relu on mu)

  // ---- VQ + concat -> z ; decoder ; student-t
  vq_concat<<<NN / 8, 256, 0, stream>>>(feat, codebook, B_mu, z);
  launch_gemm(z, P_dec, defeat, E_DIM + GH2_, D_IN, dec_b, bnd_g, bnd_b, 2, stream);
  student_t<<<NN / 8, 256, 0, stream>>>(z, cluster, qn);
}
